// TransformerBlock_46540265620291
// MI455X (gfx1250) — matrix operand order repaired and checked by reading, not compiled, not run on hardware
//
#include <hip/hip_runtime.h>
#include <hip/hip_bf16.h>

// ---------------- types ----------------
typedef __bf16 bf16_t;
typedef __attribute__((ext_vector_type(16))) __bf16 v16bf;
typedef __attribute__((ext_vector_type(8)))  float  v8f;
typedef __attribute__((ext_vector_type(4)))  unsigned int u32x4;
typedef __attribute__((ext_vector_type(8)))  int i32x8;
typedef __attribute__((ext_vector_type(4)))  int i32x4;

union Frag {
    v16bf v;
    u32x4 q[2];
};

// ---------------- constants ----------------
constexpr int B_  = 2;
constexpr int S_  = 2048;
constexpr int D_  = 1024;
constexpr int H_  = 16;
constexpr int DH_ = 64;
constexpr int M_  = B_ * S_;   // 4096
constexpr int F_  = 4 * D_;    // 4096

// ---------------- helpers ----------------
static __device__ inline bf16_t f2bf(float f) {
    unsigned u = __float_as_uint(f);
    unsigned r = u + 0x7fffu + ((u >> 16) & 1u);
    unsigned short hs = (unsigned short)(r >> 16);
    return __builtin_bit_cast(bf16_t, hs);
}

static __device__ inline float gelu_tanh_f(float x) {
    float u = 0.7978845608028654f * (x + 0.044715f * x * x * x);
    float e = __expf(2.0f * u);
    float t = 1.0f - 2.0f / (e + 1.0f);   // tanh(u), inf-safe
    return 0.5f * x * (1.0f + t);
}

static __device__ inline v8f wmma_bf16(const Frag& a, const Frag& b, v8f c) {
    return __builtin_amdgcn_wmma_f32_16x16x32_bf16(
        /*neg_a=*/false, a.v, /*neg_b=*/false, b.v,
        /*c_mod=*/(short)0, c, /*reuse_a=*/false, /*reuse_b=*/false);
}

// LDS byte offset of a shared-memory pointer: the LDS aperture keeps the
// offset in the low 32 bits of the flat address (ISA 10.2 aperture mapping).
static __device__ inline unsigned lds_off(const void* p) {
    return (unsigned)(uintptr_t)p;
}

// ---- Tensor Data Mover: load a (rows x 32) bf16 tile, row stride K elems,
// into LDS with a 48-element (96B) padded row stride. Descriptor built per
// CDNA5 ISA D# groups 0/1 (2D tile, groups 2/3 zero).
// This toolchain exposes the 6-arg builtin: (g0, g1, g2, g3, g4, cpol);
// trailing group is zero-filled.
static __device__ inline void tdm_load_tile_2d(unsigned ldsByteOff,
                                               const bf16_t* tileBase,
                                               unsigned rows,
                                               unsigned rowStrideElems) {
    unsigned long long ga = (unsigned long long)(uintptr_t)tileBase;
    u32x4 g0;
    g0[0] = 1u;                                           // count=1, user descriptor
    g0[1] = ldsByteOff;                                   // lds_addr
    g0[2] = (unsigned)ga;                                 // global_addr[31:0]
    g0[3] = (unsigned)((ga >> 32) & 0x01FFFFFFu)          // global_addr[56:32]
          | (2u << 30);                                   // type = 2 ("image")
    unsigned dim0 = rowStrideElems;                       // tensor_dim0 (OOB bound)
    unsigned dim1 = rows;                                 // tensor_dim1
    i32x8 g1;
    g1[0] = (int)((1u << 16)                              // data_size = 2 bytes
                | (1u << 20)                              // pad_enable
                | (3u << 22)                              // pad_interval: 16 DWORDs (64B row)
                | (7u << 25));                            // pad_amount: 8 DWORDs (32B pad)
    g1[1] = (int)((dim0 & 0xffffu) << 16);                // tensor_dim0[15:0]
    g1[2] = (int)(((dim0 >> 16) & 0xffffu)                // tensor_dim0[31:16]
                | ((dim1 & 0xffffu) << 16));              // tensor_dim1[15:0]
    g1[3] = (int)(((dim1 >> 16) & 0xffffu)                // tensor_dim1[31:16]
                | (32u << 16));                           // tile_dim0 = 32 elems
    g1[4] = (int)(rows & 0xffffu);                        // tile_dim1 = rows, tile_dim2 = 0
    g1[5] = (int)rowStrideElems;                          // tensor_dim0_stride[31:0]
    g1[6] = 0;                                            // stride0 hi / stride1 lo
    g1[7] = 0;
    i32x4 gz4 = {0, 0, 0, 0};
    i32x8 gz8 = {0, 0, 0, 0, 0, 0, 0, 0};
    __builtin_amdgcn_tensor_load_to_lds(g0, g1, gz4, gz4, gz8, 0);
}

// ---------------- weight convert + transpose: Wt[n*K+k] = (bf16)W[k*N+n] ----
__global__ __launch_bounds__(256) void wt_transpose(const float* __restrict__ W,
                                                    bf16_t* __restrict__ Wt,
                                                    int K, int N) {
    size_t tid = (size_t)blockIdx.x * 256 + threadIdx.x;
    if (tid >= (size_t)K * N) return;
    size_t k = tid % (size_t)K;
    size_t n = tid / (size_t)K;
    Wt[tid] = f2bf(W[k * (size_t)N + n]);
}

// ---------------- LayerNorm (fp32 in -> bf16 out), one block per row -------
__global__ __launch_bounds__(256) void layernorm_bf16(const float* __restrict__ x,
                                                      const float* __restrict__ g,
                                                      const float* __restrict__ beta,
                                                      bf16_t* __restrict__ out) {
    int row = blockIdx.x;
    int tid = threadIdx.x;
    const float* xr = x + (size_t)row * D_;
    float4 v = ((const float4*)xr)[tid];
    float s  = v.x + v.y + v.z + v.w;
    float s2 = v.x * v.x + v.y * v.y + v.z * v.z + v.w * v.w;
    for (int m = 1; m < 32; m <<= 1) {
        s  += __shfl_xor(s,  m, 32);
        s2 += __shfl_xor(s2, m, 32);
    }
    __shared__ float red[2][8];
    if ((tid & 31) == 0) { red[0][tid >> 5] = s; red[1][tid >> 5] = s2; }
    __syncthreads();
    float sum = 0.f, sum2 = 0.f;
    for (int i = 0; i < 8; ++i) { sum += red[0][i]; sum2 += red[1][i]; }
    float mu   = sum * (1.0f / D_);
    float var  = sum2 * (1.0f / D_) - mu * mu;
    float rstd = rsqrtf(var + 1e-5f);
    int base = tid * 4;
    float vv[4] = {v.x, v.y, v.z, v.w};
    for (int e = 0; e < 4; ++e) {
        int c = base + e;
        out[(size_t)row * D_ + c] = f2bf((vv[e] - mu) * rstd * g[c] + beta[c]);
    }
}

// ---------------- tiled bf16 GEMM with TDM-staged, double-buffered LDS -----
// C = A[M,K] x Bt[N,K]^T
// MODE 0: store bf16 ; MODE 1: +bias, GELU, store bf16 ; MODE 2: +bias, +residual, store f32
template <int MODE>
__global__ __launch_bounds__(256) void gemm_bf16(const bf16_t* __restrict__ A,
                                                 const bf16_t* __restrict__ Bt,
                                                 const float* __restrict__ bias,
                                                 const float* __restrict__ res,
                                                 void* __restrict__ outP,
                                                 int M, int N, int K) {
    constexpr int LDT = 48;                 // LDS K-stride = TDM 64B row + 32B pad
    __shared__ bf16_t shA[2][128 * LDT];
    __shared__ bf16_t shB[2][128 * LDT];

    int tid  = threadIdx.x;
    int lane = tid & 31;
    int wave = tid >> 5;                    // 0..7
    int l16  = lane & 15;
    int half = lane >> 4;                   // 0/1
    int m0   = blockIdx.y * 128;
    int n0   = blockIdx.x * 128;
    int wm   = (wave >> 2) * 64;            // 0/64
    int wn   = (wave & 3) * 32;             // 0/32/64/96

    v8f acc[4][2];
    for (int mt = 0; mt < 4; ++mt)
        for (int nt = 0; nt < 2; ++nt)
            for (int e = 0; e < 8; ++e) acc[mt][nt][e] = 0.0f;

    const unsigned aOff[2] = { lds_off(&shA[0][0]), lds_off(&shA[1][0]) };
    const unsigned bOff[2] = { lds_off(&shB[0][0]), lds_off(&shB[1][0]) };
    const bf16_t* aBase = A  + (size_t)m0 * K;
    const bf16_t* bBase = Bt + (size_t)n0 * K;

    int nT = K / 32;
    if (wave == 0) {                        // TDM prologue: tile 0 into buffer 0
        tdm_load_tile_2d(aOff[0], aBase, 128, (unsigned)K);
        tdm_load_tile_2d(bOff[0], bBase, 128, (unsigned)K);
    }

    for (int i = 0; i < nT; ++i) {
        int cur = i & 1;
        if (wave == 0) __builtin_amdgcn_s_wait_tensorcnt(0);  // pair(i) landed
        __syncthreads();                    // publish tile i; other buffer now free
        if (wave == 0 && i + 1 < nT) {      // DMA next tile while we compute
            tdm_load_tile_2d(aOff[1 - cur], aBase + (size_t)(i + 1) * 32, 128, (unsigned)K);
            tdm_load_tile_2d(bOff[1 - cur], bBase + (size_t)(i + 1) * 32, 128, (unsigned)K);
        }

        const bf16_t* sA = shA[cur];
        const bf16_t* sB = shB[cur];
        Frag af[4], bf[2];
        for (int mt = 0; mt < 4; ++mt) {
            const bf16_t* p = sA + (wm + mt * 16 + l16) * LDT;
            af[mt].q[0] = *(const u32x4*)(p + half * 8);        // K = h*8 .. h*8+7
            af[mt].q[1] = *(const u32x4*)(p + half * 8 + 16);   // K = h*8+16 .. +23
        }
        for (int nt = 0; nt < 2; ++nt) {
            const bf16_t* p = sB + (wn + nt * 16 + l16) * LDT;
            bf[nt].q[0] = *(const u32x4*)(p + half * 8);        // K = h*8 .. h*8+7
            bf[nt].q[1] = *(const u32x4*)(p + half * 8 + 16);   // K = h*8+16 .. +23
        }
        for (int mt = 0; mt < 4; ++mt)
            for (int nt = 0; nt < 2; ++nt)
                acc[mt][nt] = wmma_bf16(af[mt], bf[nt], acc[mt][nt]);
    }

    for (int mt = 0; mt < 4; ++mt)
        for (int nt = 0; nt < 2; ++nt)
            for (int r = 0; r < 8; ++r) {
                int gR = m0 + wm + mt * 16 + r + half * 8;
                int gC = n0 + wn + nt * 16 + l16;
                float v = acc[mt][nt][r];
                if constexpr (MODE >= 1) v += bias[gC];
                if constexpr (MODE == 1) v = gelu_tanh_f(v);
                if constexpr (MODE == 2) {
                    v += res[(size_t)gR * N + gC];
                    ((float*)outP)[(size_t)gR * N + gC] = v;
                } else {
                    ((bf16_t*)outP)[(size_t)gR * N + gC] = f2bf(v);
                }
            }
}

// ---------------- flash attention: q,k,v bf16 [B*S, H*DH] -> ctx bf16 ------
__global__ __launch_bounds__(128) void attn_kernel(const bf16_t* __restrict__ qb,
                                                   const bf16_t* __restrict__ kb,
                                                   const bf16_t* __restrict__ vb,
                                                   bf16_t* __restrict__ ctx) {
    int blk  = blockIdx.x;
    int qBlk = blk % (S_ / 64);
    int h    = (blk / (S_ / 64)) % H_;
    int b    = blk / ((S_ / 64) * H_);
    int qb0  = qBlk * 64;

    int tid  = threadIdx.x;
    int lane = tid & 31;
    int wave = tid >> 5;                    // 0..3, each wave: 16 q rows
    int l16  = lane & 15;
    int half = lane >> 4;

    __shared__ bf16_t shVt[DH_ * 32];       // V^T  [d][key], stride 32
    __shared__ bf16_t shP[4 * 16 * 32];     // per-wave P tile [row][key]

    const int    qRowBase = qb0 + wave * 16;
    const size_t headOff  = (size_t)b * S_ * D_ + (size_t)h * DH_;

    // Q fragments (2 K-steps of 32 along Dh), kept in registers
    Frag qf[2];
    {
        const bf16_t* qp = qb + headOff + (size_t)(qRowBase + l16) * D_;
        for (int ks = 0; ks < 2; ++ks) {
            int d0 = ks * 32 + half * 8;
            qf[ks].q[0] = *(const u32x4*)(qp + d0);
            qf[ks].q[1] = *(const u32x4*)(qp + d0 + 16);
        }
    }

    float mrow[8], lrow[8];
    v8f   O[4];
    for (int r = 0; r < 8; ++r) { mrow[r] = -1e30f; lrow[r] = 0.0f; }
    for (int t = 0; t < 4; ++t)
        for (int e = 0; e < 8; ++e) O[t][e] = 0.0f;

    int nKB = qb0 / 32 + 2;                 // causal: keys < qb0 + 64
    for (int kbI = 0; kbI < nKB; ++kbI) {
        int k0 = kbI * 32;

        __syncthreads();                    // protect shVt WAR
        {   // stage V^T: 4 threads per key row, 16 d each
            int key   = tid >> 2;           // 0..31
            int dBase = (tid & 3) * 16;
            const bf16_t* vp = vb + headOff + (size_t)(k0 + key) * D_ + dBase;
            u32x4 v0 = *(const u32x4*)(vp);
            u32x4 v1 = *(const u32x4*)(vp + 8);
            union { u32x4 q; unsigned short h16[8]; } t0, t1;
            t0.q = v0; t1.q = v1;
            unsigned short* sv = (unsigned short*)shVt;
            for (int e = 0; e < 8; ++e) {
                sv[(dBase + e) * 32 + key]     = t0.h16[e];
                sv[(dBase + 8 + e) * 32 + key] = t1.h16[e];
            }
        }
        __syncthreads();

        // ---- scores: S[16q x 32k] = Q(16x64) * K^T ----
        // Load all four K fragments first -> one load clause, one wait.
        Frag kf[2][2];
        for (int nt = 0; nt < 2; ++nt) {
            const bf16_t* kp = kb + headOff + (size_t)(k0 + nt * 16 + l16) * D_;
            for (int ks = 0; ks < 2; ++ks) {
                kf[nt][ks].q[0] = *(const u32x4*)(kp + ks * 32 + half * 8);
                kf[nt][ks].q[1] = *(const u32x4*)(kp + ks * 32 + half * 8 + 16);
            }
        }
        v8f sc[2];
        for (int nt = 0; nt < 2; ++nt) for (int e = 0; e < 8; ++e) sc[nt][e] = 0.0f;
        for (int nt = 0; nt < 2; ++nt)
            for (int ks = 0; ks < 2; ++ks)
                sc[nt] = wmma_bf16(qf[ks], kf[nt][ks], sc[nt]);

        // scale + causal mask
        for (int nt = 0; nt < 2; ++nt) {
            int key = k0 + nt * 16 + l16;
            for (int r = 0; r < 8; ++r) {
                float s = sc[nt][r] * 0.125f;               // 1/sqrt(64)
                int qRow = qRowBase + r + half * 8;
                sc[nt][r] = (key > qRow) ? -1e30f : s;
            }
        }
        // ---- online softmax (rows live across 16-lane halves) ----
        float p[2][8];
        for (int r = 0; r < 8; ++r) {
            float mx = fmaxf(sc[0][r], sc[1][r]);
            for (int m = 1; m < 16; m <<= 1) mx = fmaxf(mx, __shfl_xor(mx, m, 32));
            float mNew = fmaxf(mrow[r], mx);
            float corr = __expf(mrow[r] - mNew);
            mrow[r] = mNew;
            float p0 = __expf(sc[0][r] - mNew);
            float p1 = __expf(sc[1][r] - mNew);
            float rs = p0 + p1;
            for (int m = 1; m < 16; m <<= 1) rs += __shfl_xor(rs, m, 32);
            lrow[r] = lrow[r] * corr + rs;
            for (int t = 0; t < 4; ++t) O[t][r] *= corr;
            p[0][r] = p0; p[1][r] = p1;
        }
        // ---- C-layout -> A-layout via per-wave LDS round trip ----
        bf16_t* pB = shP + wave * 512;
        for (int t = 0; t < 2; ++t)
            for (int r = 0; r < 8; ++r)
                pB[(r + half * 8) * 32 + t * 16 + l16] = f2bf(p[t][r]);
        // same-wave DS ops are processed in order; keep the compiler honest:
        asm volatile("s_wait_dscnt 0x0" ::: "memory");
        Frag pf;
        {
            const bf16_t* pr = pB + l16 * 32;
            pf.q[0] = *(const u32x4*)(pr + half * 8);
            pf.q[1] = *(const u32x4*)(pr + half * 8 + 16);
        }
        // ---- ctx += P(16x32) * V(32x64) ----
        for (int t = 0; t < 4; ++t) {
            Frag vf;
            const bf16_t* vr = shVt + (t * 16 + l16) * 32;
            vf.q[0] = *(const u32x4*)(vr + half * 8);
            vf.q[1] = *(const u32x4*)(vr + half * 8 + 16);
            O[t] = wmma_bf16(pf, vf, O[t]);
        }
    }

    // normalize + store ctx (bf16, [B*S, H*DH] layout)
    for (int t = 0; t < 4; ++t)
        for (int r = 0; r < 8; ++r) {
            int row = qRowBase + r + half * 8;
            int d   = t * 16 + l16;
            ctx[headOff + (size_t)row * D_ + d] = f2bf(O[t][r] / lrow[r]);
        }
}

// ---------------- host-side orchestration ----------------
extern "C" void kernel_launch(void* const* d_in, const int* in_sizes, int n_in,
                              void* d_out, int out_size, void* d_ws, size_t ws_size,
                              hipStream_t stream) {
    (void)in_sizes; (void)n_in; (void)out_size; (void)ws_size;
    const float* x    = (const float*)d_in[0];
    const float* Wq   = (const float*)d_in[1];
    const float* Wk   = (const float*)d_in[2];
    const float* Wv   = (const float*)d_in[3];
    const float* Wo   = (const float*)d_in[4];
    const float* bo   = (const float*)d_in[5];
    const float* W1   = (const float*)d_in[6];
    const float* b1   = (const float*)d_in[7];
    const float* W2   = (const float*)d_in[8];
    const float* b2   = (const float*)d_in[9];
    const float* ln1g = (const float*)d_in[10];
    const float* ln1b = (const float*)d_in[11];
    const float* ln2g = (const float*)d_in[12];
    const float* ln2b = (const float*)d_in[13];

    char*  ws  = (char*)d_ws;
    size_t off = 0;
    auto alloc = [&](size_t bytes) -> void* {
        void* p = ws + off;
        off += (bytes + 255) & ~(size_t)255;
        return p;
    };
    bf16_t* h1  = (bf16_t*)alloc((size_t)M_ * D_ * 2);
    bf16_t* qb  = (bf16_t*)alloc((size_t)M_ * D_ * 2);
    bf16_t* kb  = (bf16_t*)alloc((size_t)M_ * D_ * 2);
    bf16_t* vb  = (bf16_t*)alloc((size_t)M_ * D_ * 2);
    bf16_t* cx  = (bf16_t*)alloc((size_t)M_ * D_ * 2);
    float*  x1  = (float*) alloc((size_t)M_ * D_ * 4);
    bf16_t* h2  = (bf16_t*)alloc((size_t)M_ * D_ * 2);
    bf16_t* mid = (bf16_t*)alloc((size_t)M_ * F_ * 2);
    bf16_t* Wqt = (bf16_t*)alloc((size_t)D_ * D_ * 2);
    bf16_t* Wkt = (bf16_t*)alloc((size_t)D_ * D_ * 2);
    bf16_t* Wvt = (bf16_t*)alloc((size_t)D_ * D_ * 2);
    bf16_t* Wot = (bf16_t*)alloc((size_t)D_ * D_ * 2);
    bf16_t* W1t = (bf16_t*)alloc((size_t)D_ * F_ * 2);
    bf16_t* W2t = (bf16_t*)alloc((size_t)F_ * D_ * 2);

    // 1) weights -> bf16, transposed to [N][K]
    int nDD = (D_ * D_ + 255) / 256;
    int nDF = (D_ * F_ + 255) / 256;
    wt_transpose<<<nDD, 256, 0, stream>>>(Wq, Wqt, D_, D_);
    wt_transpose<<<nDD, 256, 0, stream>>>(Wk, Wkt, D_, D_);
    wt_transpose<<<nDD, 256, 0, stream>>>(Wv, Wvt, D_, D_);
    wt_transpose<<<nDD, 256, 0, stream>>>(Wo, Wot, D_, D_);
    wt_transpose<<<nDF, 256, 0, stream>>>(W1, W1t, D_, F_);
    wt_transpose<<<nDF, 256, 0, stream>>>(W2, W2t, F_, D_);

    // 2) LN1
    layernorm_bf16<<<M_, 256, 0, stream>>>(x, ln1g, ln1b, h1);

    // 3) QKV projections
    dim3 gDD(D_ / 128, M_ / 128);
    gemm_bf16<0><<<gDD, 256, 0, stream>>>(h1, Wqt, nullptr, nullptr, (void*)qb, M_, D_, D_);
    gemm_bf16<0><<<gDD, 256, 0, stream>>>(h1, Wkt, nullptr, nullptr, (void*)kb, M_, D_, D_);
    gemm_bf16<0><<<gDD, 256, 0, stream>>>(h1, Wvt, nullptr, nullptr, (void*)vb, M_, D_, D_);

    // 4) flash attention
    attn_kernel<<<B_ * H_ * (S_ / 64), 128, 0, stream>>>(qb, kb, vb, cx);

    // 5) output projection + residual -> x1 (fp32)
    gemm_bf16<2><<<gDD, 256, 0, stream>>>(cx, Wot, bo, x, (void*)x1, M_, D_, D_);

    // 6) LN2
    layernorm_bf16<<<M_, 256, 0, stream>>>(x1, ln2g, ln2b, h2);

    // 7) FFN
    dim3 gDF(F_ / 128, M_ / 128);
    gemm_bf16<1><<<gDF, 256, 0, stream>>>(h2, W1t, b1, nullptr, (void*)mid, M_, F_, D_);
    gemm_bf16<2><<<gDD, 256, 0, stream>>>(mid, W2t, b2, x1, d_out, M_, D_, F_);
}
